// SimpleSSM_51883204936062
// MI455X (gfx1250) — matrix operand order repaired and checked by reading, not compiled, not run on hardware
//
#include <hip/hip_runtime.h>
#include <hip/hip_bf16.h>
#include <cstdint>
#include <cstddef>

// Problem constants (from reference): B=8, T=4096, D=1024
#define B_   8
#define T_   4096
#define D_   1024
#define MTOT (B_ * T_)          // 32768 rows for both GEMMs

// ---------- vector types ----------
typedef _Float16 h16 __attribute__((ext_vector_type(16)));
typedef _Float16 h8  __attribute__((ext_vector_type(8)));
typedef _Float16 h4  __attribute__((ext_vector_type(4)));
typedef float    f8  __attribute__((ext_vector_type(8)));
typedef float    f4  __attribute__((ext_vector_type(4)));

union H16 { h16 v; h8 h[2]; };

// =====================================================================
// f32 -> f16 conversion (vectorized, exact-size launch)
// =====================================================================
__global__ __launch_bounds__(256) void ssm_cvt_f16(const float* __restrict__ in,
                                                   _Float16* __restrict__ out,
                                                   int n4) {
  int i = blockIdx.x * 256 + threadIdx.x;
  if (i < n4) {
    f4 v = ((const f4*)in)[i];
    h4 o;
    o[0] = (_Float16)v[0]; o[1] = (_Float16)v[1];
    o[2] = (_Float16)v[2]; o[3] = (_Float16)v[3];
    ((h4*)out)[i] = o;
  }
}

// =====================================================================
// WMMA GEMM:  C[M,1024] = A[M,1024] @ W[1024,1024]^T + bias
//   - A, W in f16 row-major (W row n is output column n: einsum 'btd,ed->bte')
//   - f32 accumulate via v_wmma_f32_16x16x32_f16
//   - 128x128 block tile, BK=32, 3-stage LDS pipeline fed by
//     global_load_async_to_lds_b128 (ASYNCcnt-tracked copy engine path)
// =====================================================================
#define BK      32
#define LDA     40   // padded LDS row stride in halfs (80 B -> conflict-free b128)
#define STAGES  3

__global__ __launch_bounds__(256) void ssm_gemm_f16_wmma(
    const _Float16* __restrict__ A,
    const _Float16* __restrict__ W,
    const float*    __restrict__ bias,
    float*          __restrict__ C) {
  __shared__ __align__(16) _Float16 As[STAGES][128 * LDA];
  __shared__ __align__(16) _Float16 Bs[STAGES][128 * LDA];

  const int tid  = threadIdx.x;
  const int lane = tid & 31;
  const int wave = tid >> 5;
  const int wm   = wave & 3;   // 4 groups of 32 rows
  const int wn   = wave >> 2;  // 2 groups of 64 cols

  const int row0 = blockIdx.x * 128;   // M tile
  const int col0 = blockIdx.y * 128;   // N tile

  // staging slot: each thread owns one 32-B row slice (two b128 chunks)
  const int lr = tid >> 1;             // 0..127
  const int lc = (tid & 1) * 16;       // 0 or 16
  const _Float16* gA = A + (size_t)(row0 + lr) * 1024 + lc;
  const _Float16* gW = W + (size_t)(col0 + lr) * 1024 + lc;

  // Issue 4 async 16-B copies (A chunk0/1, B chunk0/1) into stage s for
  // K-step `kstep`.  Generic LDS pointers carry the LDS byte offset in
  // their low 32 bits (flat aperture layout), which is exactly what the
  // instruction's LDS-address VGPR wants.  `offset:16` advances BOTH the
  // global and LDS addresses, giving the contiguous second chunk.
  auto ISSUE = [&](int s, int kstep) {
    const int k0 = kstep * BK;
    unsigned la = (unsigned)(uintptr_t)&As[s][lr * LDA + lc];
    unsigned lb = (unsigned)(uintptr_t)&Bs[s][lr * LDA + lc];
    const _Float16* pa = gA + k0;
    const _Float16* pb = gW + k0;
    asm volatile(
        "global_load_async_to_lds_b128 %0, %2, off\n\t"
        "global_load_async_to_lds_b128 %0, %2, off offset:16\n\t"
        "global_load_async_to_lds_b128 %1, %3, off\n\t"
        "global_load_async_to_lds_b128 %1, %3, off offset:16"
        :: "v"(la), "v"(lb), "v"(pa), "v"(pb)
        : "memory");
  };

  f8 acc[2][4];
#pragma unroll
  for (int i = 0; i < 2; ++i)
#pragma unroll
    for (int j = 0; j < 4; ++j)
      acc[i][j] = (f8){0.f, 0.f, 0.f, 0.f, 0.f, 0.f, 0.f, 0.f};

  // per-lane fragment coordinates (this code's contiguous k order, same for A and B):
  //  A 16x32: lane m=l%16 holds contiguous K=(l/16)*16..+15
  //  B 32x16: lane col=l%16 holds contiguous K=(l/16)*16..+15
  const int ar = wm * 32 + (lane & 15);
  const int ak = (lane >> 4) * 16;
  const int br = wn * 64 + (lane & 15);
  const int bk = (lane >> 4) * 16;

  ISSUE(0, 0);
  ISSUE(1, 1);

  const int KT = 1024 / BK;  // 32
  for (int ks = 0; ks < KT; ++ks) {
    // my stage-ks copies have landed once <=4 (the next step's) remain
    if (ks + 1 < KT) asm volatile("s_wait_asynccnt 0x4" ::: "memory");
    else             asm volatile("s_wait_asynccnt 0x0" ::: "memory");
    __syncthreads();  // everyone's stage-ks data visible; stage (ks-1)%3 fully read
    if (ks + 2 < KT) ISSUE((ks + 2) % STAGES, ks + 2);

    const int cur = ks % STAGES;
    H16 af[2], bf[4];
#pragma unroll
    for (int i = 0; i < 2; ++i) {
      const _Float16* p = &As[cur][(ar + i * 16) * LDA];
      af[i].h[0] = *(const h8*)(p + ak);
      af[i].h[1] = *(const h8*)(p + ak + 8);
    }
#pragma unroll
    for (int j = 0; j < 4; ++j) {
      const _Float16* p = &Bs[cur][(br + j * 16) * LDA + bk];
      bf[j].h[0] = *(const h8*)(p);
      bf[j].h[1] = *(const h8*)(p + 8);
    }
#pragma unroll
    for (int i = 0; i < 2; ++i)
#pragma unroll
      for (int j = 0; j < 4; ++j)
        acc[i][j] = __builtin_amdgcn_wmma_f32_16x16x32_f16(
            /*neg_a=*/false, af[i].v, /*neg_b=*/false, bf[j].v,
            /*c_mod=*/(short)0, acc[i][j],
            /*reuse_a=*/false, /*reuse_b=*/false);
  }

  // epilogue: C/D layout -> VGPR r holds (m = r + (l/16)*8, n = l%16)
#pragma unroll
  for (int i = 0; i < 2; ++i) {
    const int row = row0 + wm * 32 + i * 16 + (lane >> 4) * 8;
#pragma unroll
    for (int j = 0; j < 4; ++j) {
      const int col = col0 + wn * 64 + j * 16 + (lane & 15);
      const float bv = bias[col];
      float* cp = C + (size_t)row * 1024 + col;
#pragma unroll
      for (int r = 0; r < 8; ++r)
        cp[(size_t)r * 1024] = acc[i][j][r] + bv;
    }
  }
}

// =====================================================================
// Chain-shortened sequential scan over T.
//   h_t = lam*h_{t-1} + (1-lam)*u_t, emitted every step (f16 out).
// Groups of 8: in-group prefixes depend only on that group's loads, so
// the cross-iteration dependent chain is ONE fma per 8 timesteps.
// Loads/stores coalesced across lanes (lane = channel d).
// =====================================================================
__global__ __launch_bounds__(256) void ssm_scan(const float* __restrict__ xb,
                                                const float* __restrict__ log_lambda,
                                                _Float16* __restrict__ hout) {
  const int g = blockIdx.x * 256 + threadIdx.x;  // 0..B*D-1
  const int b = g >> 10;                          // D = 1024
  const int d = g & 1023;

  const float lam = 1.f / (1.f + __expf(-__expf(log_lambda[d])));
  const float om  = 1.f - lam;
  const float l1 = lam, l2 = l1 * l1, l3 = l2 * l1, l4 = l2 * l2;
  const float l5 = l4 * l1, l6 = l4 * l2, l7 = l4 * l3, l8 = l4 * l4;

  const float* p = xb + (size_t)b * T_ * D_ + d;
  _Float16*    q = hout + (size_t)b * T_ * D_ + d;

  float h = 0.f;
  float u[8], un[8];
#pragma unroll
  for (int i = 0; i < 8; ++i) u[i] = p[(size_t)i * D_];

  const int G = T_ / 8;
  for (int gk = 0; gk < G; ++gk) {
    if (gk + 1 < G) {
      const float* pn = p + (size_t)(gk + 1) * 8 * D_;
#pragma unroll
      for (int i = 0; i < 8; ++i) un[i] = pn[(size_t)i * D_];
    }
    // two independent half-chains (depth 4), merged at depth 5
    const float a0 = om * u[0];
    const float a1 = fmaf(lam, a0, om * u[1]);
    const float a2 = fmaf(lam, a1, om * u[2]);
    const float a3 = fmaf(lam, a2, om * u[3]);
    const float c0 = om * u[4];
    const float c1 = fmaf(lam, c0, om * u[5]);
    const float c2 = fmaf(lam, c1, om * u[6]);
    const float c3 = fmaf(lam, c2, om * u[7]);
    const float p4 = fmaf(l1, a3, c0);
    const float p5 = fmaf(l2, a3, c1);
    const float p6 = fmaf(l3, a3, c2);
    const float p7 = fmaf(l4, a3, c3);
    // outputs: o_i = lam^{i+1} * h_in + prefix_i (all parallel in h)
    const float o0 = fmaf(l1, h, a0), o1 = fmaf(l2, h, a1);
    const float o2 = fmaf(l3, h, a2), o3 = fmaf(l4, h, a3);
    const float o4 = fmaf(l5, h, p4), o5 = fmaf(l6, h, p5);
    const float o6 = fmaf(l7, h, p6), o7 = fmaf(l8, h, p7);

    _Float16* qq = q + (size_t)gk * 8 * D_;
    qq[0]              = (_Float16)o0;
    qq[(size_t)1 * D_] = (_Float16)o1;
    qq[(size_t)2 * D_] = (_Float16)o2;
    qq[(size_t)3 * D_] = (_Float16)o3;
    qq[(size_t)4 * D_] = (_Float16)o4;
    qq[(size_t)5 * D_] = (_Float16)o5;
    qq[(size_t)6 * D_] = (_Float16)o6;
    qq[(size_t)7 * D_] = (_Float16)o7;

    h = o7;  // = lam^8 * h_in + p7 : single dependent fma per 8 steps
#pragma unroll
    for (int i = 0; i < 8; ++i) u[i] = un[i];
  }
}

// =====================================================================
// launch
// =====================================================================
extern "C" void kernel_launch(void* const* d_in, const int* in_sizes, int n_in,
                              void* d_out, int out_size, void* d_ws, size_t ws_size,
                              hipStream_t stream) {
  const float* x  = (const float*)d_in[0];  // [B,T,D]
  const float* ll = (const float*)d_in[1];  // [D]
  const float* bw = (const float*)d_in[2];  // [D,D]
  const float* bb = (const float*)d_in[3];  // [D]
  const float* cw = (const float*)d_in[4];  // [D,D]
  const float* cb = (const float*)d_in[5];  // [D]
  float* out = (float*)d_out;

  // workspace layout (bytes):
  //   x_h  : MTOT*D * 2 =  64 MB  (x in f16)
  //   bw_h : D*D   * 2 =    2 MB
  //   cw_h : D*D   * 2 =    2 MB
  //   xb   : MTOT*D * 4 = 128 MB  (GEMM1 output, f32; mostly L2-resident)
  //   h_h  : MTOT*D * 2 =  64 MB  (scan output, f16)
  char* ws = (char*)d_ws;
  size_t off = 0;
  _Float16* x_h  = (_Float16*)(ws + off); off += (size_t)MTOT * D_ * 2;
  _Float16* bw_h = (_Float16*)(ws + off); off += (size_t)D_ * D_ * 2;
  _Float16* cw_h = (_Float16*)(ws + off); off += (size_t)D_ * D_ * 2;
  float*    xb   = (float*)   (ws + off); off += (size_t)MTOT * D_ * 4;
  _Float16* h_h  = (_Float16*)(ws + off);

  const int n4x = MTOT * D_ / 4;  // 8,388,608
  const int n4w = D_ * D_ / 4;    //   262,144
  ssm_cvt_f16<<<n4x / 256, 256, 0, stream>>>(x,  x_h,  n4x);
  ssm_cvt_f16<<<n4w / 256, 256, 0, stream>>>(bw, bw_h, n4w);
  ssm_cvt_f16<<<n4w / 256, 256, 0, stream>>>(cw, cw_h, n4w);

  dim3 grid(MTOT / 128, D_ / 128);  // 256 x 8 blocks
  ssm_gemm_f16_wmma<<<grid, 256, 0, stream>>>(x_h, bw_h, bb, xb);
  ssm_scan<<<(B_ * D_) / 256, 256, 0, stream>>>(xb, ll, h_h);
  ssm_gemm_f16_wmma<<<grid, 256, 0, stream>>>(h_h, cw_h, cb, out);
}
